// Mozafari2018_33440615367236
// MI455X (gfx1250) — matrix operand order repaired and checked by reading, not compiled, not run on hardware
//
#include <hip/hip_runtime.h>

typedef __attribute__((ext_vector_type(16))) _Float16 v16h;
typedef __attribute__((ext_vector_type(8)))  float    v8f;

#define T_DIM   15
#define C_IN    6
#define H_IN    64
#define W_IN    64
#define FEAT    1000
#define KSZ     5
#define HO      60
#define WO      60
#define P_HW    (HO*WO)           // 3600
#define N_TOT   (T_DIM*P_HW)      // 54000
#define K_REAL  150
#define KP      160               // padded K, 5 chunks of 32
#define MP      1008              // padded M, 63 tiles of 16
#define MTILES  63
#define NTILES  3375              // 54000/16
#define NT_PER_WAVE 5
#define NGRP    (NTILES/NT_PER_WAVE)   // 675
#define NJOBS   (MTILES*NGRP)          // 42525
#define THRESH  15.0f
#define FEATURES_PER_CLASS 10
#define NEURONS (FEAT*P_HW)       // 3,600,000

// ---------------- workspace layout (bytes) ----------------
// [0,            322560)   : A packed f16  (MP*KP)
// [524288,       17804288) : B^T im2col f16 (N_TOT*KP)
// [17825792, +8)           : u64 packed argmax accumulator
// [17825800, +4)           : u32 float-bits max accumulator
#define WS_A_OFF    0
#define WS_B_OFF    (512*1024)
#define WS_SCAL_OFF (17825792)

// ---------- 1) pack weights into padded row-major f16 A ----------
__global__ void __launch_bounds__(256) pack_w(const float* __restrict__ w,
                                              _Float16* __restrict__ Ah) {
    int i = blockIdx.x * 256 + threadIdx.x;
    if (i >= MP * KP) return;
    int m = i / KP, k = i % KP;
    float v = (m < FEAT && k < K_REAL) ? w[m * K_REAL + k] : 0.0f;
    Ah[i] = (_Float16)v;
}

// ---------- 2) im2col into K-contiguous f16 B^T ----------
__global__ void __launch_bounds__(256) im2col(const float* __restrict__ x,
                                              _Float16* __restrict__ Bh) {
    int i = blockIdx.x * 256 + threadIdx.x;
    if (i >= N_TOT * KP) return;
    int n = i / KP, k = i % KP;
    float v = 0.0f;
    if (k < K_REAL) {
        int t  = n / P_HW, p = n % P_HW;
        int oy = p / WO,   ox = p % WO;
        int c  = k / (KSZ*KSZ), r = k % (KSZ*KSZ);
        int ky = r / KSZ,  kx = r % KSZ;
        v = x[((t * C_IN + c) * H_IN + (oy + ky)) * W_IN + (ox + kx)];
    }
    Bh[i] = (_Float16)v;
}

// ---------- init scalar accumulators (ws is poisoned, not re-zeroed) ----------
__global__ void init_scalars(unsigned long long* best, unsigned int* maxbits) {
    *best = 0ull;
    *maxbits = 0u;
}

// ---------- 3) WMMA GEMM + thresholding epilogue ----------
__global__ void __launch_bounds__(256) conv_wmma(const _Float16* __restrict__ Ah,
                                                 const _Float16* __restrict__ Bh,
                                                 float* __restrict__ out) {
    int wave = blockIdx.x * (256 / 32) + (threadIdx.x >> 5);
    if (wave >= NJOBS) return;               // wave-uniform: EXEC stays all-ones
    int lane  = threadIdx.x & 31;
    int lmod  = lane & 15;
    int lhalf = lane >> 4;

    int mt = wave / NGRP;
    int n0tile = (wave % NGRP) * NT_PER_WAVE;

    v8f acc[NT_PER_WAVE];
    v8f zero = {0.f,0.f,0.f,0.f,0.f,0.f,0.f,0.f};
#pragma unroll
    for (int j = 0; j < NT_PER_WAVE; ++j) acc[j] = zero;

    // A fragment: lane L holds row M = mt*16 + L%16,
    //   16 contiguous K values starting at k0 + 16*half
    const _Float16* arow = Ah + (size_t)(mt * 16 + lmod) * KP;

#pragma unroll
    for (int kc = 0; kc < KP / 32; ++kc) {
        int k0 = kc * 32;
        union { v16h v; float4 q[2]; } a;
        a.q[0] = *(const float4*)(arow + k0 + lhalf * 16);
        a.q[1] = *(const float4*)(arow + k0 + lhalf * 16 + 8);
#pragma unroll
        for (int j = 0; j < NT_PER_WAVE; ++j) {
            // B fragment: lane L holds column N = L%16,
            //   16 contiguous K values starting at k0 + 16*(L/16)
            int n = (n0tile + j) * 16 + lmod;
            const float4* bp = (const float4*)(Bh + (size_t)n * KP + k0 + lhalf * 16);
            union { v16h v; float4 q[2]; } b;
            b.q[0] = bp[0];
            b.q[1] = bp[1];
            acc[j] = __builtin_amdgcn_wmma_f32_16x16x32_f16(
                false, a.v, false, b.v, (short)0, acc[j], false, false);
        }
    }

    // Epilogue: C/D layout -> lane n = lmod, VGPR v is row m = mt*16 + v + 8*lhalf
    const size_t thr_off = (size_t)N_TOT * FEAT;   // 54,000,000
#pragma unroll
    for (int j = 0; j < NT_PER_WAVE; ++j) {
        int n = (n0tile + j) * 16 + lmod;
        int t = n / P_HW;
        int p = n % P_HW;
        size_t base = (size_t)t * FEAT * P_HW + p;
#pragma unroll
        for (int v = 0; v < 8; ++v) {
            int m = mt * 16 + v + 8 * lhalf;
            if (m < FEAT) {
                float pot  = acc[j][v];
                float thrv = (pot >= THRESH) ? pot : 0.0f;
                float spkv = (pot >= THRESH) ? 1.0f : 0.0f;
                size_t idx = base + (size_t)m * P_HW;
                out[idx] = spkv;
                out[thr_off + idx] = thrv;
            }
        }
    }
}

// ---------- per-neuron nspk / first-spike value ----------
__device__ __forceinline__ void neuron_stats(const float* __restrict__ thr,
                                             int tid, int* nspk_out, float* val_out) {
    size_t base = (size_t)tid;  // tid == f*P_HW + p == flat [F,Ho,Wo] index
    int nspk = 0;
    float vals[T_DIM];
#pragma unroll
    for (int t = 0; t < T_DIM; ++t) {
        float tv = thr[(size_t)t * FEAT * P_HW + base];
        vals[t] = tv;
        nspk += (tv > 0.0f) ? 1 : 0;
    }
    int first = T_DIM - nspk;
    if (first < 0) first = 0;
    if (first > T_DIM - 1) first = T_DIM - 1;
    float val = 0.0f;
#pragma unroll
    for (int t = 0; t < T_DIM; ++t) val = (t == first) ? vals[t] : val;  // selects, no scratch
    *nspk_out = nspk;
    *val_out = val;
}

// ---------- 4) global max of first-spike value over spiking neurons ----------
__global__ void __launch_bounds__(256) max_pass(const float* __restrict__ out,
                                                unsigned int* __restrict__ maxbits) {
    int tid = blockIdx.x * 256 + threadIdx.x;
    float cand = 0.0f;
    if (tid < NEURONS) {
        const float* thr = out + (size_t)N_TOT * FEAT;
        int nspk; float val;
        neuron_stats(thr, tid, &nspk, &val);
        cand = (nspk > 0) ? val : 0.0f;     // == max contribution of trunc = spk*vals
    }
    __shared__ float smax[256];
    smax[threadIdx.x] = cand;
    __syncthreads();
    for (int s = 128; s > 0; s >>= 1) {
        if (threadIdx.x < s)
            smax[threadIdx.x] = fmaxf(smax[threadIdx.x], smax[threadIdx.x + s]);
        __syncthreads();
    }
    if (threadIdx.x == 0)
        atomicMax(maxbits, __float_as_uint(smax[0]));  // valid: values >= 0
}

// ---------- 5) argmax of total with first-index tie-break ----------
__global__ void __launch_bounds__(256) argmax_pass(const float* __restrict__ out,
                                                   const unsigned int* __restrict__ maxbits,
                                                   unsigned long long* __restrict__ best) {
    int tid = blockIdx.x * 256 + threadIdx.x;
    unsigned long long key = 0ull;
    if (tid < NEURONS) {
        const float* thr = out + (size_t)N_TOT * FEAT;
        int nspk; float val;
        neuron_stats(thr, tid, &nspk, &val);
        float v = (float)T_DIM * __uint_as_float(*maxbits);  // trunc.max() * t
        float total = (float)nspk * (val + v);               // (trunc + spk*v).sum(axis=0)
        // total >= 0 so float bits are order-preserving; ~tid makes smaller index win ties
        key = ((unsigned long long)__float_as_uint(total) << 32) |
              (unsigned long long)(unsigned int)(~(unsigned int)tid);
    }
    __shared__ unsigned long long skey[256];
    skey[threadIdx.x] = key;
    __syncthreads();
    for (int s = 128; s > 0; s >>= 1) {
        if (threadIdx.x < s) {
            unsigned long long o = skey[threadIdx.x + s];
            if (o > skey[threadIdx.x]) skey[threadIdx.x] = o;
        }
        __syncthreads();
    }
    if (threadIdx.x == 0)
        atomicMax(best, skey[0]);
}

// ---------- 6) finalize: decode winner -> class ----------
__global__ void finalize(const unsigned long long* __restrict__ best,
                         float* __restrict__ out_cls) {
    unsigned long long k = *best;
    unsigned int idx = ~(unsigned int)(k & 0xFFFFFFFFull);
    float val = __uint_as_float((unsigned int)(k >> 32));
    int feat = (int)(idx / P_HW);
    out_cls[0] = (val != 0.0f) ? (float)(feat / FEATURES_PER_CLASS) : -1.0f;
}

extern "C" void kernel_launch(void* const* d_in, const int* in_sizes, int n_in,
                              void* d_out, int out_size, void* d_ws, size_t ws_size,
                              hipStream_t stream) {
    (void)in_sizes; (void)n_in; (void)out_size; (void)ws_size;
    const float* x = (const float*)d_in[0];   // [15,6,64,64]
    const float* w = (const float*)d_in[1];   // [1000,6,5,5]
    float* out = (float*)d_out;               // spk | thr | class

    char* ws = (char*)d_ws;
    _Float16* Ah = (_Float16*)(ws + WS_A_OFF);
    _Float16* Bh = (_Float16*)(ws + WS_B_OFF);
    unsigned long long* best = (unsigned long long*)(ws + WS_SCAL_OFF);
    unsigned int* maxbits    = (unsigned int*)(ws + WS_SCAL_OFF + 8);

    pack_w<<<(MP * KP + 255) / 256, 256, 0, stream>>>(w, Ah);
    im2col<<<(N_TOT * KP + 255) / 256, 256, 0, stream>>>(x, Bh);
    init_scalars<<<1, 1, 0, stream>>>(best, maxbits);
    conv_wmma<<<(NJOBS + 7) / 8, 256, 0, stream>>>(Ah, Bh, out);
    max_pass<<<(NEURONS + 255) / 256, 256, 0, stream>>>(out, maxbits);
    argmax_pass<<<(NEURONS + 255) / 256, 256, 0, stream>>>(out, maxbits, best);
    finalize<<<1, 1, 0, stream>>>(best, out + (size_t)2 * N_TOT * FEAT);
}
